// RelationalNet_2542620639842
// MI455X (gfx1250) — matrix operand order repaired and checked by reading, not compiled, not run on hardware
//
#include <hip/hip_runtime.h>
#include <hip/hip_bf16.h>

typedef __attribute__((ext_vector_type(16))) _Float16 v16h;
typedef __attribute__((ext_vector_type(8)))  _Float16 v8h;
typedef __attribute__((ext_vector_type(4)))  _Float16 v4h;
typedef __attribute__((ext_vector_type(8)))  float    v8f;

#define IN_DIM  512
#define OUT_DIM 512
#define BATCH   32
#define SEQ     64
#define NROWS   (BATCH * SEQ)          // 2048
#define OUT_ROWS 4096                  // 64 obj + 4032 rel per batch

// ---------------- helpers ----------------

__device__ __forceinline__ v16h mk16(const _Float16* p0, const _Float16* p1) {
    v8h lo = *(const v8h*)p0;
    v8h hi = *(const v8h*)p1;
    v16h r;
#pragma unroll
    for (int e = 0; e < 8; ++e) { r[e] = lo[e]; r[e + 8] = hi[e]; }
    return r;
}

// ---------------- kernel 0: fp32 -> f16 convert ----------------

__global__ void cvt_f32_f16(const float* __restrict__ src,
                            _Float16* __restrict__ dst, int n4) {
    int i = blockIdx.x * blockDim.x + threadIdx.x;
    if (i < n4) {
        float4 v = *(const float4*)(src + i * 4);
        v4h h;
        h[0] = (_Float16)v.x; h[1] = (_Float16)v.y;
        h[2] = (_Float16)v.z; h[3] = (_Float16)v.w;
        *(v4h*)(dst + i * 4) = h;
    }
}

// ---------------- kernel 1: L / R / obj GEMMs ----------------
// grid (128 mtiles, 2 nhalves, 3 matrices), block 128 (4 waves).
// Wave computes a 16x64 output strip with 4 WMMAs per K-step (A reused).

__global__ __launch_bounds__(128)
void gemm_lro(const _Float16* __restrict__ xh,
              const _Float16* __restrict__ Wl,
              const _Float16* __restrict__ Wr,
              const _Float16* __restrict__ Wo,
              const float* __restrict__ bl,
              const float* __restrict__ br,
              const float* __restrict__ bo,
              _Float16* __restrict__ Lbuf,
              _Float16* __restrict__ Rbuf,
              float* __restrict__ out) {
    const int lane  = threadIdx.x & 31;
    const int wave  = threadIdx.x >> 5;
    const int m0    = blockIdx.x * 16;
    const int n0    = blockIdx.y * 256 + wave * 64;
    const int mat   = blockIdx.z;
    const _Float16* W = (mat == 0) ? Wl : (mat == 1) ? Wr : Wo;

    const int rown  = lane & 15;
    const int koffA = (lane < 16) ? 0 : 8;
    const int koffB = (lane < 16) ? 0 : 8;

    v8f acc[4];
#pragma unroll
    for (int t = 0; t < 4; ++t) acc[t] = (v8f)(0.0f);

    const _Float16* arow = xh + (size_t)(m0 + rown) * IN_DIM;
    for (int k0 = 0; k0 < IN_DIM; k0 += 32) {
        const _Float16* pa = arow + k0 + koffA;
        v16h a = mk16(pa, pa + 16);
#pragma unroll
        for (int t = 0; t < 4; ++t) {
            const _Float16* pb = W + (size_t)(n0 + t * 16 + rown) * IN_DIM + k0 + koffB;
            v16h bfr = mk16(pb, pb + 16);
            acc[t] = __builtin_amdgcn_wmma_f32_16x16x32_f16(
                false, a, false, bfr, (short)0, acc[t], false, false);
        }
    }

    const int rbase = (lane < 16) ? 0 : 8;
    if (mat < 2) {
        const float* bias = (mat == 0) ? bl : br;
        _Float16* dst = (mat == 0) ? Lbuf : Rbuf;
#pragma unroll
        for (int t = 0; t < 4; ++t) {
            int col = n0 + t * 16 + rown;
            float bv = bias[col];
#pragma unroll
            for (int r = 0; r < 8; ++r) {
                int m = m0 + rbase + r;
                dst[(size_t)m * OUT_DIM + col] = (_Float16)(acc[t][r] + bv);
            }
        }
    } else {
#pragma unroll
        for (int t = 0; t < 4; ++t) {
            int col = n0 + t * 16 + rown;
            float bv = bo[col];
#pragma unroll
            for (int r = 0; r < 8; ++r) {
                int m = m0 + rbase + r;
                int b = m >> 6, s = m & 63;
                __builtin_nontemporal_store(
                    acc[t][r] + bv,
                    &out[((size_t)(b * OUT_ROWS + s)) * OUT_DIM + col]);
            }
        }
    }
}

// ---------------- kernel 2: pair build + rel GEMM ----------------
// One block per (b, i). Stage P = relu(L[b,:,:] * R[b,i,:]) in LDS (64x512 f16,
// XOR-swizzled 16B chunks). Then each of the 16 waves owns 2 DISJOINT N-tiles
// (32 cols) and iterates all 4 M-tiles, so every W_rel fragment is fetched from
// global exactly once per block (A fragments come from LDS). 8 WMMAs / K-step.

__global__ __launch_bounds__(512)
void pair_rel(const _Float16* __restrict__ Lbuf,
              const _Float16* __restrict__ Rbuf,
              const _Float16* __restrict__ Wrel,
              const float* __restrict__ brel,
              float* __restrict__ out) {
    __shared__ _Float16 P[SEQ * OUT_DIM];   // 64 KB

    const int tid = threadIdx.x;
    const int bi  = blockIdx.x;
    const int b   = bi >> 6;
    const int i   = bi & 63;

    // ---- stage A: build P ----
    {
        const _Float16* Lb   = Lbuf + (size_t)b * SEQ * OUT_DIM;
        const _Float16* Rrow = Rbuf + (size_t)(b * SEQ + i) * OUT_DIM;
        int j     = tid >> 3;          // row 0..63
        int cbase = (tid & 7) * 64;    // 8 chunks of 8 halves per thread
#pragma unroll
        for (int cc = 0; cc < 8; ++cc) {
            int c = cbase + cc * 8;
            v8h lv = *(const v8h*)(Lb + (size_t)j * OUT_DIM + c);
            v8h rv = *(const v8h*)(Rrow + c);
            v8h pv;
#pragma unroll
            for (int e = 0; e < 8; ++e) {
                float p = (float)lv[e] * (float)rv[e];
                pv[e] = (_Float16)(p > 0.0f ? p : 0.0f);
            }
            int sw = (c >> 3) ^ (j & 7);   // swizzle 16B chunks within row
            *(v8h*)(&P[j * OUT_DIM + sw * 8]) = pv;
        }
    }
    __syncthreads();

    // ---- stage B: GEMM with W_rel ----
    const int lane  = tid & 31;
    const int wave  = tid >> 5;       // 0..15
    const int n0    = wave * 32;      // each wave: cols [n0, n0+32)
    const int rown  = lane & 15;
    const int koffA = (lane < 16) ? 0 : 8;
    const int koffB = (lane < 16) ? 0 : 8;

    v8f acc[4][2];
#pragma unroll
    for (int m = 0; m < 4; ++m)
#pragma unroll
        for (int t = 0; t < 2; ++t) acc[m][t] = (v8f)(0.0f);

    for (int k0 = 0; k0 < OUT_DIM; k0 += 32) {
        // B fragments: 2 disjoint N-tiles, fetched once per block
        v16h bfr[2];
#pragma unroll
        for (int t = 0; t < 2; ++t) {
            const _Float16* pb = Wrel + (size_t)(n0 + t * 16 + rown) * OUT_DIM
                                 + k0 + koffB;
            bfr[t] = mk16(pb, pb + 16);
        }
        // A fragments from LDS: all 4 M-tiles
#pragma unroll
        for (int m = 0; m < 4; ++m) {
            int jA  = m * 16 + rown;
            int swj = jA & 7;
            int c1  = (k0 + koffA) >> 3;
            int c2  = c1 + 2;
            v16h a = mk16(&P[jA * OUT_DIM + ((c1 ^ swj) << 3)],
                          &P[jA * OUT_DIM + ((c2 ^ swj) << 3)]);
#pragma unroll
            for (int t = 0; t < 2; ++t) {
                acc[m][t] = __builtin_amdgcn_wmma_f32_16x16x32_f16(
                    false, a, false, bfr[t], (short)0, acc[m][t], false, false);
            }
        }
    }

    // ---- epilogue: bias + diagonal-skip scatter (NT stores) ----
    const int rbase = (lane < 16) ? 0 : 8;
#pragma unroll
    for (int t = 0; t < 2; ++t) {
        int col = n0 + t * 16 + rown;
        float bv = brel[col];
#pragma unroll
        for (int m = 0; m < 4; ++m) {
#pragma unroll
            for (int r = 0; r < 8; ++r) {
                int j = m * 16 + rbase + r;
                if (j != i) {
                    int orow = 64 + i * 64 + j - i - (j > i ? 1 : 0);
                    __builtin_nontemporal_store(
                        acc[m][t] [r] + bv,
                        &out[((size_t)(b * OUT_ROWS + orow)) * OUT_DIM + col]);
                }
            }
        }
    }
}

// ---------------- launcher ----------------

extern "C" void kernel_launch(void* const* d_in, const int* in_sizes, int n_in,
                              void* d_out, int out_size, void* d_ws, size_t ws_size,
                              hipStream_t stream) {
    const float* x      = (const float*)d_in[0];
    const float* W_left = (const float*)d_in[1];
    const float* b_left = (const float*)d_in[2];
    const float* W_right= (const float*)d_in[3];
    const float* b_right= (const float*)d_in[4];
    const float* W_rel  = (const float*)d_in[5];
    const float* b_rel  = (const float*)d_in[6];
    const float* W_obj  = (const float*)d_in[7];
    const float* b_obj  = (const float*)d_in[8];
    float* out = (float*)d_out;

    // workspace layout (f16 elements)
    _Float16* ws   = (_Float16*)d_ws;
    _Float16* xh   = ws;                                   // 2048*512
    _Float16* Wl   = xh + (size_t)NROWS * IN_DIM;          // 512*512
    _Float16* Wr   = Wl + (size_t)OUT_DIM * IN_DIM;
    _Float16* Wo   = Wr + (size_t)OUT_DIM * IN_DIM;
    _Float16* Wrl  = Wo + (size_t)OUT_DIM * IN_DIM;
    _Float16* Lbuf = Wrl + (size_t)OUT_DIM * OUT_DIM;      // 2048*512
    _Float16* Rbuf = Lbuf + (size_t)NROWS * OUT_DIM;       // 2048*512

    // convert fp32 -> f16
    {
        int n4 = (NROWS * IN_DIM) / 4;
        cvt_f32_f16<<<(n4 + 255) / 256, 256, 0, stream>>>(x, xh, n4);
        int w4 = (OUT_DIM * IN_DIM) / 4;
        cvt_f32_f16<<<(w4 + 255) / 256, 256, 0, stream>>>(W_left,  Wl,  w4);
        cvt_f32_f16<<<(w4 + 255) / 256, 256, 0, stream>>>(W_right, Wr,  w4);
        cvt_f32_f16<<<(w4 + 255) / 256, 256, 0, stream>>>(W_obj,   Wo,  w4);
        cvt_f32_f16<<<(w4 + 255) / 256, 256, 0, stream>>>(W_rel,   Wrl, w4);
    }

    // L / R / obj GEMMs
    {
        dim3 grid(NROWS / 16, 2, 3);
        gemm_lro<<<grid, 128, 0, stream>>>(xh, Wl, Wr, Wo,
                                           b_left, b_right, b_obj,
                                           Lbuf, Rbuf, out);
    }

    // pair + rel GEMM
    {
        pair_rel<<<BATCH * SEQ, 512, 0, stream>>>(Lbuf, Rbuf, Wrl, b_rel, out);
    }
}
